// MultiHeadAttention_9079560864541
// MI455X (gfx1250) — matrix operand order repaired and checked by reading, not compiled, not run on hardware
//
#include <hip/hip_runtime.h>
#include <hip/hip_bf16.h>

typedef __attribute__((ext_vector_type(16))) _Float16 v16h;
typedef __attribute__((ext_vector_type(8)))  _Float16 v8h;
typedef __attribute__((ext_vector_type(4)))  _Float16 v4h;
typedef __attribute__((ext_vector_type(8)))  float    v8f;

// Build a 16-half fragment from two 16B (8-half) chunks.
__device__ __forceinline__ v16h ld_frag(const _Float16* p0, const _Float16* p1) {
    v8h lo = *(const v8h*)p0;
    v8h hi = *(const v8h*)p1;
    v16h r;
#pragma unroll
    for (int i = 0; i < 8; ++i) { r[i] = lo[i]; r[i + 8] = hi[i]; }
    return r;
}

__device__ __forceinline__ v8f zero8() {
    v8f z;
#pragma unroll
    for (int i = 0; i < 8; ++i) z[i] = 0.0f;
    return z;
}

__device__ __forceinline__ v8f wmma_f16(v16h a, v16h b, v8f c) {
    // D = A(16x32 f16) * B(32x16 f16) + C(16x16 f32)
    return __builtin_amdgcn_wmma_f32_16x16x32_f16(
        /*neg_a=*/false, a, /*neg_b=*/false, b,
        /*c_mod=*/(short)0, c, /*reuse_a=*/false, /*reuse_b=*/false);
}

// ---------------------------------------------------------------------------
// f32 -> f16 convert, 4 elements/thread
// ---------------------------------------------------------------------------
__global__ __launch_bounds__(256)
void cvt_f16_kernel(const float* __restrict__ in, _Float16* __restrict__ out, int n) {
    int i = (blockIdx.x * 256 + threadIdx.x) * 4;
    if (i < n) {
        float4 f = *(const float4*)(in + i);
        v4h h;
        h[0] = (_Float16)f.x; h[1] = (_Float16)f.y;
        h[2] = (_Float16)f.z; h[3] = (_Float16)f.w;
        *(v4h*)(out + i) = h;
    }
}

// ---------------------------------------------------------------------------
// WMMA GEMM: Y[m][n] = sum_k A16[m][k] * W16[n][k] + bias[n]
//   M = 4096 (B*S), N = 1024 (D), K = 1024 (D)
//   Block: 128 threads = 4 waves; block tile 64(M) x 128(N);
//   each wave: 32x64 = 2x4 grid of 16x16 wmma accumulators, K-step 32.
// MODE: 1 = Q (scale 0.125, store f16 [b][h][s][dk])
//       2 = K (store f16 [b][h][s][dk])
//       3 = V (store f16 transposed [b][h][dk][s])
//       4 = final (store f32 row-major [m][n])
// ---------------------------------------------------------------------------
template<int MODE>
__global__ __launch_bounds__(128)
void gemm_wmma_kernel(const _Float16* __restrict__ A, const _Float16* __restrict__ W,
                      const float* __restrict__ bias, void* __restrict__ outp,
                      int Kdim) {
    const int lane = threadIdx.x & 31;
    const int wave = threadIdx.x >> 5;
    const int lm = lane & 15;           // column / row-within-16
    const int lh = lane >> 4;           // half-wave select
    const int mbase = blockIdx.x * 64 + (wave & 1) * 32;
    const int nbase = blockIdx.y * 128 + (wave >> 1) * 64;

    // A-fragment row pointers (lane = row, K chunks at lh*8 and lh*8+16)
    const _Float16* arow0 = A + (size_t)(mbase + lm) * Kdim + lh * 8;
    const _Float16* arow1 = arow0 + (size_t)16 * Kdim;
    // B-fragment pointers: lane = output column, K chunks at lh*8 and lh*8+16
    const _Float16* brow[4];
#pragma unroll
    for (int j = 0; j < 4; ++j)
        brow[j] = W + (size_t)(nbase + j * 16 + lm) * Kdim + lh * 8;

    v8f acc[2][4];
#pragma unroll
    for (int i = 0; i < 2; ++i)
#pragma unroll
        for (int j = 0; j < 4; ++j) acc[i][j] = zero8();

    for (int k0 = 0; k0 < Kdim; k0 += 32) {
        v16h a0 = ld_frag(arow0 + k0, arow0 + k0 + 16);
        v16h a1 = ld_frag(arow1 + k0, arow1 + k0 + 16);
        v16h b[4];
#pragma unroll
        for (int j = 0; j < 4; ++j)
            b[j] = ld_frag(brow[j] + k0, brow[j] + k0 + 16);
#pragma unroll
        for (int j = 0; j < 4; ++j) {
            acc[0][j] = wmma_f16(a0, b[j], acc[0][j]);
            acc[1][j] = wmma_f16(a1, b[j], acc[1][j]);
        }
    }

    // Epilogue. C layout: VGPR r -> row (lh*8 + r), lane lm -> column.
#pragma unroll
    for (int j = 0; j < 4; ++j) {
        const int n = nbase + j * 16 + lm;
        const float bj = bias[n];
#pragma unroll
        for (int i = 0; i < 2; ++i) {
#pragma unroll
            for (int r = 0; r < 8; ++r) {
                const int m = mbase + i * 16 + lh * 8 + r;
                float v = acc[i][j][r] + bj;
                if (MODE == 4) {
                    ((float*)outp)[(size_t)m * 1024 + n] = v;
                } else {
                    const int bb = m >> 11, s = m & 2047;   // S = 2048
                    const int h  = n >> 6,  dk = n & 63;    // DK = 64
                    if (MODE == 1) v *= 0.125f;             // 1/sqrt(64)
                    size_t idx;
                    if (MODE == 3)
                        idx = ((size_t)(bb * 16 + h) * 64 + dk) * 2048 + s;   // V^T
                    else
                        idx = ((size_t)(bb * 16 + h) * 2048 + s) * 64 + dk;   // Q, K
                    ((_Float16*)outp)[idx] = (_Float16)v;
                }
            }
        }
    }
}

// ---------------------------------------------------------------------------
// Flash attention: Q pre-scaled; K [b][h][s][dk]; V transposed [b][h][dk][s].
// Block = 4 waves; block owns (b, h, 64 query rows); wave owns 16 query rows.
// Streams 64-key tiles: 8 wmma for scores, online softmax, 8 wmma for P*V.
// Writes ctx f16 as [b][s][h*64+dk] so it feeds the output projection GEMM.
// ---------------------------------------------------------------------------
__global__ __launch_bounds__(128)
void attn_kernel(const _Float16* __restrict__ Q, const _Float16* __restrict__ K,
                 const _Float16* __restrict__ Vt, const int* __restrict__ mask,
                 _Float16* __restrict__ ctx) {
    __shared__ _Float16 plds[4][16 * 64];   // per-wave P staging (C-layout -> A-layout)

    const int lane = threadIdx.x & 31;
    const int wave = threadIdx.x >> 5;
    const int lm = lane & 15, lh = lane >> 4;
    const int b = blockIdx.z, h = blockIdx.y;
    const int q0 = blockIdx.x * 64 + wave * 16;
    const size_t bh = (size_t)(b * 16 + h);

    const _Float16* Qb = Q  + bh * 2048 * 64;
    const _Float16* Kb = K  + bh * 2048 * 64;
    const _Float16* Vb = Vt + bh * 64 * 2048;
    const int* mb = mask + b * 2048;

    // Persistent Q A-fragments (16 rows x 64 dk = 2 fragments)
    const _Float16* qrow = Qb + (size_t)(q0 + lm) * 64 + lh * 8;
    const v16h qa0 = ld_frag(qrow,      qrow + 16);
    const v16h qa1 = ld_frag(qrow + 32, qrow + 48);

    v8f cacc[4];
    float mrun[8], lrun[8];
#pragma unroll
    for (int t = 0; t < 4; ++t) cacc[t] = zero8();
#pragma unroll
    for (int r = 0; r < 8; ++r) { mrun[r] = -1e30f; lrun[r] = 0.0f; }

    _Float16* pw = plds[wave];

    for (int kt = 0; kt < 2048; kt += 64) {
        // ---- scores: 4 column tiles of 16 keys, K-dim = 64 (2 wmma each)
        v8f s[4];
#pragma unroll
        for (int t = 0; t < 4; ++t) {
            const _Float16* kr = Kb + (size_t)(kt + t * 16 + lm) * 64 + lh * 8;
            v16h kb0 = ld_frag(kr,      kr + 16);   // dk 0..31
            v16h kb1 = ld_frag(kr + 32, kr + 48);   // dk 32..63
            v8f sc = zero8();
            sc = wmma_f16(qa0, kb0, sc);
            sc = wmma_f16(qa1, kb1, sc);
            const float mbias = (mb[kt + t * 16 + lm] == 0) ? -1.0e9f : 0.0f;
#pragma unroll
            for (int r = 0; r < 8; ++r) sc[r] += mbias;
            s[t] = sc;
        }

        // ---- online softmax (row stats; reduce over 16-lane C-matrix halves)
        float mnew[8], corr[8];
#pragma unroll
        for (int r = 0; r < 8; ++r) {
            float mx = fmaxf(fmaxf(s[0][r], s[1][r]), fmaxf(s[2][r], s[3][r]));
#pragma unroll
            for (int off = 1; off < 16; off <<= 1)
                mx = fmaxf(mx, __shfl_xor(mx, off, 32));
            mnew[r] = fmaxf(mrun[r], mx);
            corr[r] = __expf(mrun[r] - mnew[r]);
            mrun[r] = mnew[r];
        }
#pragma unroll
        for (int t = 0; t < 4; ++t) {
#pragma unroll
            for (int r = 0; r < 8; ++r) {
                float p = __expf(s[t][r] - mnew[r]);
                s[t][r] = p;
                pw[(lh * 8 + r) * 64 + t * 16 + lm] = (_Float16)p;
            }
        }
#pragma unroll
        for (int r = 0; r < 8; ++r) {
            float rs = (s[0][r] + s[1][r]) + (s[2][r] + s[3][r]);
#pragma unroll
            for (int off = 1; off < 16; off <<= 1)
                rs += __shfl_xor(rs, off, 32);
            lrun[r] = lrun[r] * corr[r] + rs;
        }
#pragma unroll
        for (int t = 0; t < 4; ++t)
#pragma unroll
            for (int r = 0; r < 8; ++r) cacc[t][r] *= corr[r];

        // ---- P (A-frags via LDS, same-wave DS ordering) x V^T (B-frags)
        const _Float16* prow = pw + lm * 64 + lh * 8;
        v16h pa0 = ld_frag(prow,      prow + 16);   // keys 0..31 of tile
        v16h pa1 = ld_frag(prow + 32, prow + 48);   // keys 32..63
#pragma unroll
        for (int t = 0; t < 4; ++t) {
            const _Float16* vr = Vb + (size_t)(t * 16 + lm) * 2048 + kt + lh * 8;
            v16h vb0 = ld_frag(vr,      vr + 16);
            v16h vb1 = ld_frag(vr + 32, vr + 48);
            cacc[t] = wmma_f16(pa0, vb0, cacc[t]);
            cacc[t] = wmma_f16(pa1, vb1, cacc[t]);
        }
    }

    // ---- epilogue: normalize, store ctx f16 as [b][s][h*64+dk]
#pragma unroll
    for (int r = 0; r < 8; ++r) lrun[r] = 1.0f / lrun[r];
#pragma unroll
    for (int t = 0; t < 4; ++t) {
#pragma unroll
        for (int r = 0; r < 8; ++r) {
            const int qq = q0 + lh * 8 + r;
            const int dk = t * 16 + lm;
            ctx[((size_t)(b * 2048) + qq) * 1024 + h * 64 + dk] =
                (_Float16)(cacc[t][r] * lrun[r]);
        }
    }
}

// ---------------------------------------------------------------------------
extern "C" void kernel_launch(void* const* d_in, const int* in_sizes, int n_in,
                              void* d_out, int out_size, void* d_ws, size_t ws_size,
                              hipStream_t stream) {
    (void)in_sizes; (void)n_in; (void)out_size; (void)ws_size;
    const float* x    = (const float*)d_in[0];
    const int*   mask = (const int*)d_in[1];
    const float* Wq   = (const float*)d_in[2];
    const float* bq   = (const float*)d_in[3];
    const float* Wk   = (const float*)d_in[4];
    const float* bk   = (const float*)d_in[5];
    const float* Wv   = (const float*)d_in[6];
    const float* bv   = (const float*)d_in[7];
    const float* Wo   = (const float*)d_in[8];
    const float* bo   = (const float*)d_in[9];
    float* out = (float*)d_out;

    char* ws = (char*)d_ws;
    _Float16* xh   = (_Float16*)(ws);                      // 8 MB  [4096][1024]
    _Float16* Wq16 = (_Float16*)(ws + 8388608);            // 2 MB each
    _Float16* Wk16 = Wq16 + 1048576;
    _Float16* Wv16 = Wk16 + 1048576;
    _Float16* Wo16 = Wv16 + 1048576;
    _Float16* Q16  = (_Float16*)(ws + 16777216);           // [b][h][s][dk], scaled
    _Float16* K16  = (_Float16*)(ws + 25165824);           // [b][h][s][dk]
    _Float16* Vt16 = (_Float16*)(ws + 33554432);           // [b][h][dk][s]
    _Float16* C16  = (_Float16*)(ws + 41943040);           // [4096][1024]

    cvt_f16_kernel<<<4096, 256, 0, stream>>>(x,  xh,   4194304);
    cvt_f16_kernel<<<1024, 256, 0, stream>>>(Wq, Wq16, 1048576);
    cvt_f16_kernel<<<1024, 256, 0, stream>>>(Wk, Wk16, 1048576);
    cvt_f16_kernel<<<1024, 256, 0, stream>>>(Wv, Wv16, 1048576);
    cvt_f16_kernel<<<1024, 256, 0, stream>>>(Wo, Wo16, 1048576);

    dim3 g(64, 8), blk(128);
    gemm_wmma_kernel<1><<<g, blk, 0, stream>>>(xh, Wq16, bq, Q16,  1024);
    gemm_wmma_kernel<2><<<g, blk, 0, stream>>>(xh, Wk16, bk, K16,  1024);
    gemm_wmma_kernel<3><<<g, blk, 0, stream>>>(xh, Wv16, bv, Vt16, 1024);

    attn_kernel<<<dim3(32, 16, 2), 128, 0, stream>>>(Q16, K16, Vt16, mask, C16);

    gemm_wmma_kernel<4><<<g, blk, 0, stream>>>(C16, Wo16, bo, out, 1024);
}
